// WindowAttention_5669356836345
// MI455X (gfx1250) — matrix operand order repaired and checked by reading, not compiled, not run on hardware
//
#include <hip/hip_runtime.h>

typedef __attribute__((ext_vector_type(16))) __bf16 v16bf;
typedef __attribute__((ext_vector_type(8)))  float  v8f;

#define B_TOT   1024
#define NTOK    49
#define CDIM    384
#define NHEAD   12
#define HD      32
#define NWIN    64
#define QKVDIM  1152
#define MROWS   (B_TOT * NTOK)   // 50176

union FragB16 {
  v16bf   v;
  unsigned u[8];
  __bf16  h[16];
};

__device__ __forceinline__ v8f zero8() {
  v8f z = {0.f,0.f,0.f,0.f,0.f,0.f,0.f,0.f};
  return z;
}

// 16-bit A-fragment: element pair j (elements 2j,2j+1) -> starting K index
__device__ __forceinline__ int a_pair_k(int j, int hi) {
  return 16*hi + 2*j;
}

// ---------------- prep kernels ----------------
__global__ void cvt_f32_bf16(const float* __restrict__ in, __bf16* __restrict__ out, int n) {
  int i = blockIdx.x * blockDim.x + threadIdx.x;
  int stride = gridDim.x * blockDim.x;
  for (; i < n; i += stride) out[i] = (__bf16)in[i];
}

// bm[w][h][i][j] = (mask!=0) ? mask : table[pos_idx][h]
__global__ void biasmask_kernel(const float* __restrict__ mask, const int* __restrict__ pos_idx,
                                const float* __restrict__ table, float* __restrict__ bm) {
  int idx = blockIdx.x * blockDim.x + threadIdx.x;
  const int total = NWIN * NHEAD * NTOK * NTOK;
  if (idx >= total) return;
  int j = idx % NTOK;
  int t = idx / NTOK;
  int i = t % NTOK;  t /= NTOK;
  int h = t % NHEAD;
  int w = t / NHEAD;
  float mv = mask[(w * NTOK + i) * NTOK + j];
  float out = (mv != 0.0f) ? mv : table[pos_idx[(w * NTOK + i) * NTOK + j] * NHEAD + h];
  bm[idx] = out;
}

// ---------------- QKV projection GEMM ----------------
// qkv[row, col] = x[row,:] . w_qkv[col,:] + b_qkv[col]; scatter to q/k/v [B_,H,49,32] bf16
__global__ __launch_bounds__(32)
void qkv_gemm(const float* __restrict__ x, const __bf16* __restrict__ w,
              const float* __restrict__ bqkv,
              __bf16* __restrict__ qg, __bf16* __restrict__ kg, __bf16* __restrict__ vg) {
  const int lane = threadIdx.x & 31;
  const int hi   = lane >> 4;
  const int l15  = lane & 15;
  const int row0 = blockIdx.x * 16;
  const int n0   = blockIdx.y * 64;

  v8f acc[4];
  #pragma unroll
  for (int t = 0; t < 4; ++t) acc[t] = zero8();

  const int row = row0 + l15;
  const float* xr = x + (size_t)row * CDIM;

  for (int k0 = 0; k0 < CDIM; k0 += 32) {
    FragB16 a;
    #pragma unroll
    for (int j = 0; j < 8; ++j) {
      int k = k0 + a_pair_k(j, hi);
      float2 f = *(const float2*)(xr + k);
      union { unsigned u; __bf16 h[2]; } p;
      p.h[0] = (__bf16)f.x; p.h[1] = (__bf16)f.y;
      a.u[j] = p.u;
    }
    #pragma unroll
    for (int t = 0; t < 4; ++t) {
      const __bf16* wr = w + (size_t)(n0 + t*16 + l15) * CDIM + k0 + hi*16;
      FragB16 b;
      #pragma unroll
      for (int j = 0; j < 8; ++j) b.u[j] = *(const unsigned*)(wr + 2*j);
      acc[t] = __builtin_amdgcn_wmma_f32_16x16x32_bf16(false, a.v, false, b.v,
                                                       (short)0, acc[t], false, false);
    }
  }

  const float scale = 0.17677669529663687f;  // 1/sqrt(32)
  #pragma unroll
  for (int t = 0; t < 4; ++t) {
    int col   = n0 + t*16 + l15;
    int which = col / CDIM;
    int rem   = col - which * CDIM;
    int h = rem >> 5, d = rem & 31;
    float bias = bqkv[col];
    __bf16* dst = (which == 0) ? qg : (which == 1) ? kg : vg;
    #pragma unroll
    for (int e = 0; e < 8; ++e) {
      int r   = row0 + e + 8*hi;
      int b_  = r / NTOK;
      int tok = r - b_ * NTOK;
      float val = acc[t][e] + bias;
      if (which == 0) val *= scale;
      dst[(((size_t)b_ * NHEAD + h) * NTOK + tok) * HD + d] = (__bf16)val;
    }
  }
}

// ---------------- fused attention per (window, head) ----------------
__global__ __launch_bounds__(32)
void attn_kernel(const __bf16* __restrict__ qg, const __bf16* __restrict__ kg,
                 const __bf16* __restrict__ vg, const float* __restrict__ bm,
                 __bf16* __restrict__ ctx) {
  __shared__ float P[64][65];

  const int lane = threadIdx.x & 31;
  const int hi   = lane >> 4;
  const int l15  = lane & 15;
  const int b_   = blockIdx.x;
  const int h    = blockIdx.y;
  const int w    = b_ & (NWIN - 1);   // window index = b_ % 64

  const size_t bh = (size_t)b_ * NHEAD + h;
  const __bf16* qrow = qg + bh * NTOK * HD;
  const __bf16* krow = kg + bh * NTOK * HD;
  const __bf16* vrow = vg + bh * NTOK * HD;

  // ---- Q as A-fragments (rows padded 49->64 with zeros) ----
  FragB16 aq[4];
  #pragma unroll
  for (int i = 0; i < 4; ++i) {
    int m = i*16 + l15;
    if (m < NTOK) {
      #pragma unroll
      for (int j = 0; j < 8; ++j)
        aq[i].u[j] = *(const unsigned*)(qrow + m*HD + a_pair_k(j, hi));
    } else {
      #pragma unroll
      for (int j = 0; j < 8; ++j) aq[i].u[j] = 0u;
    }
  }

  // ---- K^T as B-fragments (lane = key token column, elem -> d = 16*hi + e) ----
  FragB16 bk[4];
  #pragma unroll
  for (int t = 0; t < 4; ++t) {
    int tt = t*16 + l15;
    if (tt < NTOK) {
      #pragma unroll
      for (int j = 0; j < 8; ++j)
        bk[t].u[j] = *(const unsigned*)(krow + tt*HD + 16*hi + 2*j);
    } else {
      #pragma unroll
      for (int j = 0; j < 8; ++j) bk[t].u[j] = 0u;
    }
  }

  // ---- S = Q K^T : 4x4 tiles ----
  v8f s[4][4];
  #pragma unroll
  for (int i = 0; i < 4; ++i)
    #pragma unroll
    for (int j = 0; j < 4; ++j)
      s[i][j] = __builtin_amdgcn_wmma_f32_16x16x32_bf16(false, aq[i].v, false, bk[j].v,
                                                        (short)0, zero8(), false, false);

  // ---- bias+mask, pad columns, rowwise softmax ----
  const float* bmrow = bm + ((size_t)w * NHEAD + h) * NTOK * NTOK;
  #pragma unroll
  for (int i = 0; i < 4; ++i) {
    #pragma unroll
    for (int e = 0; e < 8; ++e) {
      int m = i*16 + 8*hi + e;
      #pragma unroll
      for (int j = 0; j < 4; ++j) {
        int n2 = j*16 + l15;
        float v = s[i][j][e];
        if (n2 >= NTOK)     v = -1e30f;
        else if (m < NTOK)  v += bmrow[m * NTOK + n2];
        s[i][j][e] = v;
      }
      float mx = fmaxf(fmaxf(s[i][0][e], s[i][1][e]), fmaxf(s[i][2][e], s[i][3][e]));
      #pragma unroll
      for (int off = 1; off < 16; off <<= 1) mx = fmaxf(mx, __shfl_xor(mx, off, 32));
      float sum = 0.f;
      #pragma unroll
      for (int j = 0; j < 4; ++j) {
        float p = __expf(s[i][j][e] - mx);
        s[i][j][e] = p;
        sum += p;
      }
      #pragma unroll
      for (int off = 1; off < 16; off <<= 1) sum += __shfl_xor(sum, off, 32);
      float inv = 1.0f / sum;
      #pragma unroll
      for (int j = 0; j < 4; ++j) s[i][j][e] *= inv;
    }
  }

  // ---- re-swizzle P (D-layout -> A-layout) through LDS ----
  #pragma unroll
  for (int i = 0; i < 4; ++i)
    #pragma unroll
    for (int j = 0; j < 4; ++j)
      #pragma unroll
      for (int e = 0; e < 8; ++e)
        P[i*16 + 8*hi + e][j*16 + l15] = s[i][j][e];
  __syncthreads();

  // ---- O = P V : accumulate over 2 K-steps of 32 ----
  v8f o[4][2];
  #pragma unroll
  for (int i = 0; i < 4; ++i)
    #pragma unroll
    for (int jn = 0; jn < 2; ++jn) o[i][jn] = zero8();

  for (int ks = 0; ks < 2; ++ks) {
    FragB16 bv[2];
    #pragma unroll
    for (int jn = 0; jn < 2; ++jn) {
      int d = jn*16 + l15;
      #pragma unroll
      for (int e = 0; e < 16; ++e) {
        int tok = ks*32 + 16*hi + e;
        bv[jn].h[e] = (tok < NTOK) ? vrow[tok*HD + d] : (__bf16)0.0f;
      }
    }
    #pragma unroll
    for (int i = 0; i < 4; ++i) {
      FragB16 ap;
      int m = i*16 + l15;
      #pragma unroll
      for (int e = 0; e < 16; ++e) {
        int j  = e >> 1;
        int kk = ks*32 + a_pair_k(j, hi) + (e & 1);
        ap.h[e] = (__bf16)P[m][kk];
      }
      #pragma unroll
      for (int jn = 0; jn < 2; ++jn)
        o[i][jn] = __builtin_amdgcn_wmma_f32_16x16x32_bf16(false, ap.v, false, bv[jn].v,
                                                           (short)0, o[i][jn], false, false);
    }
  }

  // ---- store ctx[b_, tok, h*32+d] (bf16) ----
  #pragma unroll
  for (int i = 0; i < 4; ++i)
    #pragma unroll
    for (int jn = 0; jn < 2; ++jn)
      #pragma unroll
      for (int e = 0; e < 8; ++e) {
        int m = i*16 + 8*hi + e;
        if (m < NTOK)
          ctx[((size_t)b_ * NTOK + m) * CDIM + h*HD + jn*16 + l15] = (__bf16)o[i][jn][e];
      }
}

// ---------------- output projection GEMM ----------------
__global__ __launch_bounds__(32)
void proj_gemm(const __bf16* __restrict__ A, const __bf16* __restrict__ w,
               const float* __restrict__ bias, float* __restrict__ out) {
  const int lane = threadIdx.x & 31;
  const int hi   = lane >> 4;
  const int l15  = lane & 15;
  const int row0 = blockIdx.x * 16;
  const int n0   = blockIdx.y * 64;

  v8f acc[4];
  #pragma unroll
  for (int t = 0; t < 4; ++t) acc[t] = zero8();

  const __bf16* ar = A + (size_t)(row0 + l15) * CDIM;

  for (int k0 = 0; k0 < CDIM; k0 += 32) {
    FragB16 a;
    #pragma unroll
    for (int j = 0; j < 8; ++j)
      a.u[j] = *(const unsigned*)(ar + k0 + a_pair_k(j, hi));
    #pragma unroll
    for (int t = 0; t < 4; ++t) {
      const __bf16* wr = w + (size_t)(n0 + t*16 + l15) * CDIM + k0 + hi*16;
      FragB16 b;
      #pragma unroll
      for (int j = 0; j < 8; ++j) b.u[j] = *(const unsigned*)(wr + 2*j);
      acc[t] = __builtin_amdgcn_wmma_f32_16x16x32_bf16(false, a.v, false, b.v,
                                                       (short)0, acc[t], false, false);
    }
  }

  #pragma unroll
  for (int t = 0; t < 4; ++t) {
    int col = n0 + t*16 + l15;
    float bv = bias[col];
    #pragma unroll
    for (int e = 0; e < 8; ++e) {
      int r = row0 + e + 8*hi;
      out[(size_t)r * CDIM + col] = acc[t][e] + bv;
    }
  }
}

// ---------------- host launch ----------------
extern "C" void kernel_launch(void* const* d_in, const int* in_sizes, int n_in,
                              void* d_out, int out_size, void* d_ws, size_t ws_size,
                              hipStream_t stream) {
  (void)in_sizes; (void)n_in; (void)out_size; (void)ws_size;

  const float* x       = (const float*)d_in[0];
  const float* mask    = (const float*)d_in[1];
  const int*   pos_idx = (const int*)d_in[2];
  const float* table   = (const float*)d_in[3];
  const float* w_qkv   = (const float*)d_in[4];
  const float* b_qkv   = (const float*)d_in[5];
  const float* w_proj  = (const float*)d_in[6];
  const float* b_proj  = (const float*)d_in[7];
  float* out = (float*)d_out;

  char* ws = (char*)d_ws;
  size_t off = 0;
  auto alloc = [&](size_t bytes) -> void* {
    off = (off + 255) & ~(size_t)255;
    void* p = ws + off;
    off += bytes;
    return p;
  };

  const size_t qkv_elems  = (size_t)QKVDIM * CDIM;       // 442368
  const size_t proj_elems = (size_t)CDIM * CDIM;         // 147456
  const size_t bm_elems   = (size_t)NWIN * NHEAD * NTOK * NTOK;  // 1843968
  const size_t hstate     = (size_t)B_TOT * NHEAD * NTOK * HD;   // 19267584
  const size_t ctx_elems  = (size_t)MROWS * CDIM;        // 19267584

  __bf16* wqkvb  = (__bf16*)alloc(qkv_elems * 2);
  __bf16* wprojb = (__bf16*)alloc(proj_elems * 2);
  float*  bmbuf  = (float*)alloc(bm_elems * 4);
  __bf16* qg     = (__bf16*)alloc(hstate * 2);
  __bf16* kg     = (__bf16*)alloc(hstate * 2);
  __bf16* vg     = (__bf16*)alloc(hstate * 2);
  __bf16* ctx    = (__bf16*)alloc(ctx_elems * 2);

  cvt_f32_bf16<<<(qkv_elems + 255) / 256, 256, 0, stream>>>(w_qkv, wqkvb, (int)qkv_elems);
  cvt_f32_bf16<<<(proj_elems + 255) / 256, 256, 0, stream>>>(w_proj, wprojb, (int)proj_elems);
  biasmask_kernel<<<(bm_elems + 255) / 256, 256, 0, stream>>>(mask, pos_idx, table, bmbuf);

  qkv_gemm<<<dim3(MROWS / 16, QKVDIM / 64), 32, 0, stream>>>(x, wqkvb, b_qkv, qg, kg, vg);
  attn_kernel<<<dim3(B_TOT, NHEAD), 32, 0, stream>>>(qg, kg, vg, bmbuf, ctx);
  proj_gemm<<<dim3(MROWS / 16, CDIM / 64), 32, 0, stream>>>(ctx, wprojb, b_proj, out);
}
